// Generator_27195732918380
// MI455X (gfx1250) — matrix operand order repaired and checked by reading, not compiled, not run on hardware
//
#include <hip/hip_runtime.h>

// ---------------------------------------------------------------------------
// Types for CDNA5 WMMA (wave32): D(f32 16x16) = A(bf16 16x32) x B(bf16 32x16) + C
// ---------------------------------------------------------------------------
typedef __attribute__((ext_vector_type(16))) __bf16 bf16x16;
typedef __attribute__((ext_vector_type(8)))  __bf16 bf16x8;
typedef __attribute__((ext_vector_type(8)))  float  f32x8;

#define LSTM_HID 256
#define LSTM_GATES 1024

static __device__ __forceinline__ bf16x16 cat8(bf16x8 lo, bf16x8 hi)
{
    return __builtin_shufflevector(lo, hi, 0,1,2,3,4,5,6,7,8,9,10,11,12,13,14,15);
}

// ---------------------------------------------------------------------------
// Batched GEMM:  C[b] = A[b] (MxK bf16, row-major) * Wt[b] (KxN bf16, K-major
// == W^T) + bias.  Requirements (guaranteed by the host code below):
//   K % 64 == 0, M % (64*MT) == 0, gridDim.x*16*NT covers N exactly.
// One block = 4 waves; each wave owns a (16*MT) x (16*NT) register block,
// K unrolled by 2.  Branch-free inner loop; every B fragment is reused by MT
// WMMAs and every A fragment by NT WMMAs.
// ---------------------------------------------------------------------------
template <int MT, int NT>
__global__ void __launch_bounds__(128)
wmma_gemm_bf16_t(const __bf16* __restrict__ A, long long strideA,
                 const __bf16* __restrict__ Wt, long long strideW,
                 const float* __restrict__ bias,
                 float* __restrict__ C, long long strideC,
                 int M, int N, int K)
{
    const int bz = blockIdx.z;
    A  += (long long)bz * strideA;
    Wt += (long long)bz * strideW;
    C  += (long long)bz * strideC;

    const int lane = threadIdx.x & 31;
    const int wave = threadIdx.x >> 5;
    const int m0   = blockIdx.y * (64 * MT) + wave * 16;   // mt tile at m0+64*mt
    const int n0b  = blockIdx.x * (16 * NT);
    if (m0 >= M) return;                 // uniform per wave, no barriers below

    f32x8 acc[MT][NT];
#pragma unroll
    for (int mt = 0; mt < MT; ++mt)
#pragma unroll
        for (int tn = 0; tn < NT; ++tn) acc[mt][tn] = f32x8{};

    // A fragment addressing (16-bit A 16x32 layout): lanes 0-15 carry K
    // {kb..kb+7, 16+kb..16+kb+7} with kb=0, lanes 16-31 with kb=8.
    const int mrow  = m0 + (lane & 15);
    const int kbase = (lane >> 4) * 8;
    // B fragment addressing: lane carries column N = lane&15, K rows
    // {kb..kb+7, 16+kb..16+kb+7} (same order as A).
    const int kB = (lane >> 4) * 8;

    const __bf16* aRow[MT];
#pragma unroll
    for (int mt = 0; mt < MT; ++mt)
        aRow[mt] = A + (long long)(mrow + 64 * mt) * K + kbase;
    const __bf16* bRow = Wt + (long long)kB * N + n0b + (lane & 15);
    const long long bStep32 = 32LL * N;

    for (int kt = 0; kt < K; kt += 64) {
        // ---- load all fragments for this K-slab ----
        bf16x16 a0[MT], a1[MT];
#pragma unroll
        for (int mt = 0; mt < MT; ++mt) {
            const __bf16* ap = aRow[mt] + kt;
            bf16x8 lo0 = *(const bf16x8*)(ap);
            bf16x8 hi0 = *(const bf16x8*)(ap + 16);
            bf16x8 lo1 = *(const bf16x8*)(ap + 32);
            bf16x8 hi1 = *(const bf16x8*)(ap + 48);
            a0[mt] = cat8(lo0, hi0);
            a1[mt] = cat8(lo1, hi1);
        }
        const __bf16* bp0 = bRow + (long long)kt * N;
        const __bf16* bp1 = bp0 + bStep32;
        bf16x16 b0[NT], b1[NT];
#pragma unroll
        for (int tn = 0; tn < NT; ++tn) for (int s = 0; s < 16; ++s) b0[tn][s] = bp0[(long long)(s < 8 ? s : s + 8) * N + tn * 16];
#pragma unroll
        for (int tn = 0; tn < NT; ++tn) for (int s = 0; s < 16; ++s) b1[tn][s] = bp1[(long long)(s < 8 ? s : s + 8) * N + tn * 16];

        // ---- 2*MT*NT WMMAs; same-acc chains separated by MT*NT-1 others ----
#pragma unroll
        for (int mt = 0; mt < MT; ++mt)
#pragma unroll
            for (int tn = 0; tn < NT; ++tn)
                acc[mt][tn] = __builtin_amdgcn_wmma_f32_16x16x32_bf16(
                    false, a0[mt], false, b0[tn], (short)0, acc[mt][tn],
                    false, false);
#pragma unroll
        for (int mt = 0; mt < MT; ++mt)
#pragma unroll
            for (int tn = 0; tn < NT; ++tn)
                acc[mt][tn] = __builtin_amdgcn_wmma_f32_16x16x32_bf16(
                    false, a1[mt], false, b1[tn], (short)0, acc[mt][tn],
                    false, false);
    }

    // C/D layout: VGPR r -> row m0 + r + 8*(lane>=16), col n0 + (lane&15)
    const int colb   = lane & 15;
    const int rowsel = (lane >> 4) * 8;
#pragma unroll
    for (int mt = 0; mt < MT; ++mt) {
#pragma unroll
        for (int tn = 0; tn < NT; ++tn) {
            const int col = n0b + tn * 16 + colb;
            const float bv = bias ? bias[col] : 0.0f;
            float* cp = C + (long long)(m0 + 64 * mt + rowsel) * N + col;
#pragma unroll
            for (int r = 0; r < 8; ++r)
                cp[(long long)r * N] = acc[mt][tn][r] + bv;
        }
    }
}

// ---------------------------------------------------------------------------
// Elementwise helpers
// ---------------------------------------------------------------------------
__global__ void convert_pad_bf16_k(const float* __restrict__ src,
                                   __bf16* __restrict__ dst,
                                   long long M, int K, int Kp)
{
    long long idx = (long long)blockIdx.x * blockDim.x + threadIdx.x;
    long long total = M * (long long)Kp;
    if (idx >= total) return;
    long long m = idx / Kp;
    int k = (int)(idx - m * Kp);
    dst[idx] = (k < K) ? (__bf16)src[m * (long long)K + k] : (__bf16)0.0f;
}

// src: batch x N x K (f32 row-major)  ->  dst: batch x Kp x N (bf16, zero pad)
__global__ void transpose_to_bf16_k(const float* __restrict__ src,
                                    __bf16* __restrict__ dst,
                                    int N, int K, int Kp,
                                    long long sStride, long long dStride)
{
    int bz = blockIdx.z;
    long long idx = (long long)blockIdx.x * blockDim.x + threadIdx.x;
    long long total = (long long)Kp * N;
    if (idx >= total) return;
    int k = (int)(idx / N);
    int n = (int)(idx - (long long)k * N);
    __bf16 v = (__bf16)0.0f;
    if (k < K) v = (__bf16)src[bz * sStride + (long long)n * K + k];
    dst[bz * dStride + idx] = v;
}

__global__ void bias_add_k(const float* __restrict__ b1,
                           const float* __restrict__ b2,
                           float* __restrict__ out, int n)
{
    int i = blockIdx.x * blockDim.x + threadIdx.x;
    if (i < n) out[i] = b1[i] + b2[i];
}

// dst[i,t,:] = src[i, clip(len_i-1-t, 0, T-1), :]   (bf16 rows of width Kp)
__global__ void reverse_rows_bf16_k(const __bf16* __restrict__ src,
                                    __bf16* __restrict__ dst,
                                    const int* __restrict__ lens,
                                    int T, int Kp, long long total)
{
    long long idx = (long long)blockIdx.x * blockDim.x + threadIdx.x;
    if (idx >= total) return;
    long long row = idx / Kp;
    int k = (int)(idx - row * Kp);
    long long i = row / T;
    int t = (int)(row - i * T);
    int ts = lens[i] - 1 - t;
    ts = ts < 0 ? 0 : (ts >= T ? T - 1 : ts);
    dst[idx] = src[(i * T + ts) * (long long)Kp + k];
}

// One LSTM time step (pointwise): gates = pre[:,t,:] + hh; update h,c; emit
// masked output and bf16 copy of h for the next recurrence GEMM.
__global__ void lstm_cell_k(const float* __restrict__ pre,
                            const float* __restrict__ hh,
                            float* __restrict__ h, float* __restrict__ c,
                            __bf16* __restrict__ hb,
                            const int* __restrict__ lens,
                            int t, int T,
                            float* __restrict__ out, int out_ld,
                            long long NH)
{
    long long idx = (long long)blockIdx.x * blockDim.x + threadIdx.x;
    if (idx >= NH) return;
    long long i = idx >> 8;           // seq index (HID == 256)
    int j = (int)(idx & 255);
    const float* p = pre + (i * T + t) * (long long)LSTM_GATES;
    const float* g = hh + i * (long long)LSTM_GATES;
    float ig = p[j]       + g[j];
    float fg = p[256 + j] + g[256 + j];
    float gg = p[512 + j] + g[512 + j];
    float og = p[768 + j] + g[768 + j];
    ig = 1.0f / (1.0f + expf(-ig));
    fg = 1.0f / (1.0f + expf(-fg));
    og = 1.0f / (1.0f + expf(-og));
    gg = tanhf(gg);
    float c_old = c[idx], h_old = h[idx];
    float c_new = fg * c_old + ig * gg;
    float h_new = og * tanhf(c_new);
    bool valid = t < lens[i];
    float hk = valid ? h_new : h_old;
    float ck = valid ? c_new : c_old;
    h[idx] = hk;
    c[idx] = ck;
    hb[idx] = (__bf16)hk;
    out[i * (long long)T * out_ld + (long long)t * out_ld + j] = valid ? h_new : 0.0f;
}

// Un-reverse bwd-direction outputs into hseq[:, :, 256:512] with masking.
__global__ void unreverse_k(const float* __restrict__ tmp,
                            float* __restrict__ hseq,
                            const int* __restrict__ lens,
                            int T, long long total)
{
    long long idx = (long long)blockIdx.x * blockDim.x + threadIdx.x;
    if (idx >= total) return;
    long long i = idx / ((long long)T * 256);
    long long rem = idx - i * T * 256;
    int t = (int)(rem >> 8);
    int j = (int)(rem & 255);
    int len = lens[i];
    float v = 0.0f;
    if (t < len) v = tmp[(i * T + (len - 1 - t)) * 256LL + j];
    hseq[(i * T + t) * 512LL + 256 + j] = v;
}

// Last-valid-state pooling with item-count masking: enc[b,item,:]
__global__ void pool_last_k(const float* __restrict__ hseq,
                            const int* __restrict__ name_len,
                            const int* __restrict__ seq_len,
                            float* __restrict__ enc,
                            int max_items, int T, long long total)
{
    long long idx = (long long)blockIdx.x * blockDim.x + threadIdx.x;
    if (idx >= total) return;
    long long n = idx >> 9;           // /512
    int j = (int)(idx & 511);
    long long b = n / max_items;
    int item = (int)(n - b * max_items);
    float v = 0.0f;
    if (item < seq_len[b]) {
        int t = name_len[n] - 1;
        if (t < 0) t = 0;
        v = hseq[(n * T + t) * 512LL + j];
    }
    enc[idx] = v;
}

// col_enc[b,c,:] += tab_enc[b, col_to_tab[b,c], :]  where c < col_len[b]
__global__ void add_tab_k(float* __restrict__ col_enc,
                          const float* __restrict__ tab_enc,
                          const int* __restrict__ col_to_tab,
                          const int* __restrict__ col_len,
                          long long total)
{
    long long idx = (long long)blockIdx.x * blockDim.x + threadIdx.x;
    if (idx >= total) return;
    long long bc = idx >> 9;
    int j = (int)(idx & 511);
    long long b = bc >> 6;            // /64
    int cidx = (int)(bc & 63);
    if (cidx < col_len[b]) {
        int tt = col_to_tab[b * 64 + cidx];
        col_enc[idx] += tab_enc[(b * 16 + tt) * 512LL + j];
    }
}

// ---------------------------------------------------------------------------
// Host driver
// ---------------------------------------------------------------------------
static inline unsigned blocks_for(long long total, int bs)
{
    return (unsigned)((total + bs - 1) / bs);
}

extern "C" void kernel_launch(void* const* d_in, const int* in_sizes, int n_in,
                              void* d_out, int out_size, void* d_ws, size_t ws_size,
                              hipStream_t stream)
{
    (void)in_sizes; (void)n_in; (void)out_size; (void)ws_size;

    const int B = 128, MAX_Q = 64, MAX_COL = 64, MAX_TAB = 16, MAX_NAME = 8;
    const int NWORD = 300, NWORD_P = 320;   // padded to multiple of 64

    auto fin = [&](int i) { return (const float*)d_in[i]; };
    auto iin = [&](int i) { return (const int*)d_in[i]; };

    const float* q_emb    = fin(0);
    const int*   q_len    = iin(1);
    const float* col_emb  = fin(2);
    const int*   col_nlen = iin(3);
    const int*   col_len  = iin(4);
    const float* tab_emb  = fin(5);
    const int*   tab_nlen = iin(6);
    const int*   tab_len  = iin(7);
    const int*   col2tab  = iin(8);
    const int QP = 9, CP = 25;              // q_lstm_params / ct_lstm_params bases
    const float* att_W = fin(41);
    const float* att_b = fin(42);

    float* dout = (float*)d_out;

    // ---- workspace bump allocator (256B aligned) ----
    char* W = (char*)d_ws;
    size_t off = 0;
    auto alloc = [&](size_t bytes) -> char* {
        off = (off + 255) & ~(size_t)255;
        char* p = W + off;
        off += bytes;
        return p;
    };

    // Transposed bf16 weights + combined biases: [lstm][layer][dir]
    __bf16* wihT[2][2][2];
    __bf16* whhT[2][2][2];
    float*  bcv [2][2][2];
    for (int L = 0; L < 2; ++L)
        for (int l = 0; l < 2; ++l)
            for (int d = 0; d < 2; ++d) {
                int base = (L == 0 ? QP : CP) + l * 8 + d * 4;
                int K  = (l == 0) ? NWORD   : 512;
                int Kp = (l == 0) ? NWORD_P : 512;
                wihT[L][l][d] = (__bf16*)alloc((size_t)Kp * 1024 * sizeof(__bf16));
                transpose_to_bf16_k<<<dim3(blocks_for((long long)Kp * 1024, 256), 1, 1),
                                      256, 0, stream>>>(
                    fin(base + 0), wihT[L][l][d], 1024, K, Kp, 0, 0);
                whhT[L][l][d] = (__bf16*)alloc((size_t)256 * 1024 * sizeof(__bf16));
                transpose_to_bf16_k<<<dim3(blocks_for(256LL * 1024, 256), 1, 1),
                                      256, 0, stream>>>(
                    fin(base + 1), whhT[L][l][d], 1024, 256, 256, 0, 0);
                bcv[L][l][d] = (float*)alloc(1024 * sizeof(float));
                bias_add_k<<<4, 256, 0, stream>>>(fin(base + 2), fin(base + 3),
                                                  bcv[L][l][d], 1024);
            }
    __bf16* attWT = (__bf16*)alloc((size_t)512 * 512 * sizeof(__bf16));
    transpose_to_bf16_k<<<dim3(blocks_for(512LL * 512, 256), 1, 1), 256, 0, stream>>>(
        att_W, attWT, 512, 512, 512, 0, 0);

    // Persistent intermediates
    float*  col_enc = (float*)alloc((size_t)B * 64 * 512 * sizeof(float));
    float*  tab_enc = (float*)alloc((size_t)B * 16 * 512 * sizeof(float));
    float*  atted   = (float*)alloc((size_t)B * 64 * 512 * sizeof(float));
    __bf16* attedb  = (__bf16*)alloc((size_t)B * 64 * 512 * sizeof(__bf16));
    __bf16* colWt   = (__bf16*)alloc((size_t)B * 512 * 64 * sizeof(__bf16));
    __bf16* tabWt   = (__bf16*)alloc((size_t)B * 512 * 16 * sizeof(__bf16));

    // Reusable scratch, sized for the column path (Nseq=8192, T=8, M=65536)
    const long long MMAX = 65536, NSEQMAX = 8192;
    __bf16* xb    = (__bf16*)alloc((size_t)MMAX * 512 * sizeof(__bf16));
    __bf16* xrev  = (__bf16*)alloc((size_t)MMAX * 512 * sizeof(__bf16));
    float*  pre   = (float*)alloc((size_t)MMAX * 1024 * sizeof(float));
    float*  hseq  = (float*)alloc((size_t)MMAX * 512 * sizeof(float));
    float*  tmpb  = (float*)alloc((size_t)MMAX * 256 * sizeof(float));
    float*  gates = (float*)alloc((size_t)NSEQMAX * 1024 * sizeof(float));
    float*  hbuf  = (float*)alloc((size_t)NSEQMAX * 256 * sizeof(float));
    float*  cbuf  = (float*)alloc((size_t)NSEQMAX * 256 * sizeof(float));
    __bf16* hb    = (__bf16*)alloc((size_t)NSEQMAX * 256 * sizeof(__bf16));

    // ---- 2-layer bidirectional LSTM over sequences already staged in xb ----
    auto run_bilstm = [&](long long Nseq, int T, const int* lens, int Lw, int K0p) {
        long long M = Nseq * T;
        for (int l = 0; l < 2; ++l) {
            int Kp = (l == 0) ? K0p : 512;
            if (l == 1) {
                long long tot = M * 512;
                convert_pad_bf16_k<<<blocks_for(tot, 256), 256, 0, stream>>>(
                    hseq, xb, M, 512, 512);
            }
            for (int d = 0; d < 2; ++d) {
                const __bf16* xin = xb;
                if (d == 1) {
                    long long tot = M * (long long)Kp;
                    reverse_rows_bf16_k<<<blocks_for(tot, 256), 256, 0, stream>>>(
                        xb, xrev, lens, T, Kp, tot);
                    xin = xrev;
                }
                // pre = x @ Wih^T + (bih + bhh)    -- one big WMMA GEMM
                wmma_gemm_bf16_t<2, 4><<<dim3(1024 / 64, (unsigned)(M / 128), 1),
                                         128, 0, stream>>>(
                    xin, 0, wihT[Lw][l][d], 0, bcv[Lw][l][d],
                    pre, 0, (int)M, 1024, Kp);
                hipMemsetAsync(hbuf, 0, (size_t)Nseq * 256 * sizeof(float), stream);
                hipMemsetAsync(cbuf, 0, (size_t)Nseq * 256 * sizeof(float), stream);
                hipMemsetAsync(hb,   0, (size_t)Nseq * 256 * sizeof(__bf16), stream);
                for (int t = 0; t < T; ++t) {
                    // gates_hh = h @ Whh^T    -- recurrence WMMA GEMM
                    wmma_gemm_bf16_t<2, 4><<<dim3(1024 / 64,
                                                  (unsigned)(Nseq / 128), 1),
                                             128, 0, stream>>>(
                        hb, 0, whhT[Lw][l][d], 0, nullptr,
                        gates, 0, (int)Nseq, 1024, 256);
                    long long nh = Nseq * 256;
                    lstm_cell_k<<<blocks_for(nh, 256), 256, 0, stream>>>(
                        pre, gates, hbuf, cbuf, hb, lens, t, T,
                        d == 0 ? hseq : tmpb, d == 0 ? 512 : 256, nh);
                }
                if (d == 1) {
                    long long tot = M * 256;
                    unreverse_k<<<blocks_for(tot, 256), 256, 0, stream>>>(
                        tmpb, hseq, lens, T, tot);
                }
            }
        }
    };

    // ---- column name encoder: 8192 seqs x T=8 ----
    {
        long long M = (long long)B * MAX_COL * MAX_NAME;      // 65536
        convert_pad_bf16_k<<<blocks_for(M * NWORD_P, 256), 256, 0, stream>>>(
            col_emb, xb, M, NWORD, NWORD_P);
        run_bilstm((long long)B * MAX_COL, MAX_NAME, col_nlen, 1, NWORD_P);
        long long tot = (long long)B * MAX_COL * 512;
        pool_last_k<<<blocks_for(tot, 256), 256, 0, stream>>>(
            hseq, col_nlen, col_len, col_enc, MAX_COL, MAX_NAME, tot);
    }

    // ---- table name encoder: 2048 seqs x T=8 ----
    {
        long long M = (long long)B * MAX_TAB * MAX_NAME;      // 16384
        convert_pad_bf16_k<<<blocks_for(M * NWORD_P, 256), 256, 0, stream>>>(
            tab_emb, xb, M, NWORD, NWORD_P);
        run_bilstm((long long)B * MAX_TAB, MAX_NAME, tab_nlen, 1, NWORD_P);
        long long tot = (long long)B * MAX_TAB * 512;
        pool_last_k<<<blocks_for(tot, 256), 256, 0, stream>>>(
            hseq, tab_nlen, tab_len, tab_enc, MAX_TAB, MAX_NAME, tot);
    }

    // ---- col_enc += gather(tab_enc, col_to_tab) * col_valid ----
    {
        long long tot = (long long)B * MAX_COL * 512;
        add_tab_k<<<blocks_for(tot, 256), 256, 0, stream>>>(
            col_enc, tab_enc, col2tab, col_len, tot);
    }

    // ---- question encoder: 128 seqs x T=64; then atted = q_enc @ attW^T + b ----
    {
        long long M = (long long)B * MAX_Q;                   // 8192
        convert_pad_bf16_k<<<blocks_for(M * NWORD_P, 256), 256, 0, stream>>>(
            q_emb, xb, M, NWORD, NWORD_P);
        run_bilstm((long long)B, MAX_Q, q_len, 0, NWORD_P);
        convert_pad_bf16_k<<<blocks_for(M * 512, 256), 256, 0, stream>>>(
            hseq, xb, M, 512, 512);
        wmma_gemm_bf16_t<2, 4><<<dim3(512 / 64, (unsigned)(M / 128), 1),
                                 128, 0, stream>>>(
            xb, 0, attWT, 0, att_b, atted, 0, (int)M, 512, 512);
        convert_pad_bf16_k<<<blocks_for(M * 512, 256), 256, 0, stream>>>(
            atted, attedb, M, 512, 512);
    }

    // ---- per-batch transposes of col_enc / tab_enc into K-major bf16 ----
    transpose_to_bf16_k<<<dim3(blocks_for(512LL * 64, 256), 1, B), 256, 0, stream>>>(
        col_enc, colWt, 64, 512, 512, 64LL * 512, 512LL * 64);
    transpose_to_bf16_k<<<dim3(blocks_for(512LL * 16, 256), 1, B), 256, 0, stream>>>(
        tab_enc, tabWt, 16, 512, 512, 16LL * 512, 512LL * 16);

    // ---- batched attention GEMMs straight into d_out ----
    // col_att[b] = atted[b] (64x512) @ col_enc[b]^T (512x64)  -> M=64: MT=1,NT=4
    wmma_gemm_bf16_t<1, 4><<<dim3(1, 1, B), 128, 0, stream>>>(
        attedb, 64LL * 512, colWt, 512LL * 64, nullptr,
        dout, 64LL * 64, 64, 64, 512);
    // tab_att[b] = atted[b] (64x512) @ tab_enc[b]^T (512x16)  -> M=64: MT=1,NT=1
    wmma_gemm_bf16_t<1, 1><<<dim3(1, 1, B), 128, 0, stream>>>(
        attedb, 64LL * 512, tabWt, 512LL * 16, nullptr,
        dout + (long long)B * 64 * 64, 64LL * 16, 64, 16, 512);
}
